// MambaBranchEncoder_42047729828174
// MI455X (gfx1250) — matrix operand order repaired and checked by reading, not compiled, not run on hardware
//
#include <hip/hip_runtime.h>
#include <hip/hip_bf16.h>
#include <math.h>

// ---------------------------------------------------------------------------
// MambaBranchEncoder forward for MI455X (gfx1250, wave32, WMMA).
// GEMMs: bf16 WMMA (v_wmma_f32_16x16x32_bf16), f32 accumulate.
// Weight tiles are double-buffered in LDS via global_load_async_to_lds_b128:
// the copy for K-step k+1 is in flight (ASYNCcnt) while step k computes.
// ---------------------------------------------------------------------------

typedef __bf16 bf16_t;
typedef __attribute__((ext_vector_type(16))) __bf16 v16bf;
typedef __attribute__((ext_vector_type(8)))  __bf16 v8bf;
typedef __attribute__((ext_vector_type(8)))  float  v8f;

#define DEV __device__ __forceinline__

// generic pointer -> wave-relative LDS byte offset (addrspace(3) ptrtoint)
typedef __attribute__((address_space(3))) void lds_void;
DEV unsigned lds_offset_of(const void* p) {
    return (unsigned)(unsigned long long)(lds_void*)p;
}

// async 16B memory -> LDS copy, per active lane (tracked on ASYNCcnt)
DEV void async_load_b128(unsigned lds, unsigned long long gaddr) {
    asm volatile("global_load_async_to_lds_b128 %0, %1, off"
                 :: "v"(lds), "v"(gaddr) : "memory");
}
DEV void wait_async0() {
    asm volatile("s_wait_asynccnt 0" ::: "memory");
}

// ---- WMMA fragment loaders (ISA 7.12.2 layouts, wave32) -------------------
// A-matrix 16x32 (MxK), 16-bit: lane l (l<16): row M=l, elems 0..7 = K[0..7],
// elems 8..15 = K[16..23]; lane 16+l: K[8..15], K[24..31].
DEV v16bf load_frag_a(const bf16_t* __restrict__ A, int lda, int rm, int k0, int lane) {
    int r = rm + (lane & 15);
    int chunk = (lane >> 4) * 8;
    const bf16_t* p = A + (size_t)r * lda + k0 + chunk;
    v8bf lo = *(const v8bf*)p;
    v8bf hi = *(const v8bf*)(p + 16);
    v16bf f;
#pragma unroll
    for (int i = 0; i < 8; ++i) { f[i] = lo[i]; f[i + 8] = hi[i]; }
    return f;
}

// B-matrix 32x16 (KxN), 16-bit, read from the LDS-staged tile sB[64][32]:
// lane l (l<16): col N=l, elems 0..7 = K[0..7], elems 8..15 = K[16..23]; lane 16+l: K[8..15], K[24..31].
DEV v16bf load_frag_b_lds(const bf16_t* sB, int n0, int lane) {
    int n  = n0 + (lane & 15);
    int kb = (lane >> 4) << 3;
    const bf16_t* p = sB + n * 32 + kb;
    v8bf lo = *(const v8bf*)p;
    v8bf hi = *(const v8bf*)(p + 16);
    v16bf f;
#pragma unroll
    for (int i = 0; i < 8; ++i) { f[i] = lo[i]; f[i + 8] = hi[i]; }
    return f;
}

DEV float act_apply(float v, int act) {
    if (act == 1) return 0.5f * v * (1.0f + erff(v * 0.70710678118654752f)); // exact gelu
    if (act == 2) return (v > 15.f) ? v : log1pf(__expf(v));                 // softplus
    return v;
}

// C/D 16x16 f32: lane l<16: VGPR v -> (M=v, N=l); lane 16+l: (M=v+8, N=l).
DEV void store_frag(const v8f& c, int r0, int n0, int lane,
                    const float* __restrict__ bias,
                    float* __restrict__ outf, bf16_t* __restrict__ outb,
                    int N, int act, int smode, int Lseq) {
    int col = n0 + (lane & 15);
    int rb  = r0 + ((lane >> 4) << 3);
    float bv = bias ? bias[col] : 0.f;
#pragma unroll
    for (int v = 0; v < 8; ++v) {
        int row = rb + v;
        float val = act_apply(c[v] + bv, act);
        int orow = row;
        if (smode == 2) {                   // add with L-reversal (backward branch)
            int bb = row / Lseq, tt = row % Lseq;
            orow = bb * Lseq + (Lseq - 1 - tt);
        }
        size_t idx = (size_t)orow * N + col;
        if (smode == 0) {
            if (outf) outf[idx] = val;
            if (outb) outb[idx] = (bf16_t)val;
        } else {                            // smode 1/2: accumulate into outf
            outf[idx] += val;
        }
    }
}

// ---- GEMM: out[M,N] = act(A[M,K] * W[N,K]^T + bias) -----------------------
// 128 threads = 4 waves stacked in M; wave tile = 32(M)x64(N) (2x4 frags).
// Block tile = 128x64.  B tile (64 N-rows x 32 K, 4KB) is shared by the 4
// waves and double-buffered in LDS: async copy of the next tile overlaps
// the WMMAs of the current tile.
__global__ __launch_bounds__(128)
void k_gemm_bf16(const bf16_t* __restrict__ A, int lda,
                 const bf16_t* __restrict__ W, int ldw,
                 const float* __restrict__ bias,
                 float* __restrict__ outf, bf16_t* __restrict__ outb,
                 int N, int K, int act, int smode, int Lseq) {
    __shared__ bf16_t sB[2][64 * 32];       // two 4 KB [n][k] tiles

    const int lane = threadIdx.x & 31;
    const int wave = threadIdx.x >> 5;
    const int rm = blockIdx.y * 128 + wave * 32;
    const int nb = blockIdx.x * 64;

    // stage one 64x32 B tile (256 16B chunks, 2 per thread) into sB[buf]
    auto stage = [&](int buf, int k0) {
        int chunk = threadIdx.x;
#pragma unroll
        for (int c2 = 0; c2 < 2; ++c2, chunk += 128) {
            int row = chunk >> 2;                   // 0..63
            int kc  = (chunk & 3) << 3;             // 0,8,16,24 elems
            const bf16_t* g = W + (size_t)(nb + row) * ldw + (k0 + kc);
            async_load_b128(lds_offset_of(&sB[buf][row * 32 + kc]),
                            (unsigned long long)(size_t)g);
        }
    };

    stage(0, 0);                            // prologue: tile 0 in flight

    v8f c0[4] = {}, c1[4] = {};
    for (int k0 = 0; k0 < K; k0 += 32) {
        const int buf = (k0 >> 5) & 1;
        // issue this step's A-fragment global loads before the sync point
        v16bf a0 = load_frag_a(A, lda, rm,      k0, lane);
        v16bf a1 = load_frag_a(A, lda, rm + 16, k0, lane);

        wait_async0();                      // this wave's copies into sB[buf] done
        __syncthreads();                    // tile visible; prev-buffer reads done
        if (k0 + 32 < K) stage(buf ^ 1, k0 + 32);   // prefetch next tile (async)

#pragma unroll
        for (int ni = 0; ni < 4; ++ni) {
            v16bf bf = load_frag_b_lds(&sB[buf][0], ni * 16, lane);
            c0[ni] = __builtin_amdgcn_wmma_f32_16x16x32_bf16(false, a0, false, bf, (short)0, c0[ni], false, false);
            c1[ni] = __builtin_amdgcn_wmma_f32_16x16x32_bf16(false, a1, false, bf, (short)0, c1[ni], false, false);
        }
    }
#pragma unroll
    for (int ni = 0; ni < 4; ++ni) {
        store_frag(c0[ni], rm,      nb + ni * 16, lane, bias, outf, outb, N, act, smode, Lseq);
        store_frag(c1[ni], rm + 16, nb + ni * 16, lane, bias, outf, outb, N, act, smode, Lseq);
    }
}

// ---- Elementwise helpers ---------------------------------------------------
__global__ void k_f32_to_bf16(const float* __restrict__ in, bf16_t* __restrict__ out, long n) {
    long i = (long)blockIdx.x * blockDim.x + threadIdx.x;
    if (i < n) out[i] = (bf16_t)in[i];
}

// reversed-in-L f32 -> bf16 (feeds backward Mamba branch)
__global__ void k_rev_to_bf16(const float* __restrict__ in, bf16_t* __restrict__ out,
                              int Lseq, int D, long n) {
    long i = (long)blockIdx.x * blockDim.x + threadIdx.x;
    if (i >= n) return;
    int d = (int)(i % D);
    long r = i / D;
    int t = (int)(r % Lseq);
    long b = r / Lseq;
    out[(b * Lseq + (Lseq - 1 - t)) * (long)D + d] = (bf16_t)in[i];
}

// depthwise causal conv (K taps) + SiLU; reads xc half of xz (row stride 2*DI)
__global__ void k_conv_silu(const float* __restrict__ xz,
                            const float* __restrict__ cw, const float* __restrict__ cb,
                            float* __restrict__ xc, bf16_t* __restrict__ xcb,
                            int Lseq, int DIc, int KC, long n) {
    long i = (long)blockIdx.x * blockDim.x + threadIdx.x;
    if (i >= n) return;
    int d = (int)(i % DIc);
    long bt = i / DIc;
    int t = (int)(bt % Lseq);
    long b = bt / Lseq;
    float acc = cb[d];
    for (int k = 0; k < KC; ++k) {
        int tt = t + k - (KC - 1);
        if (tt >= 0) acc += cw[d * KC + k] * xz[(b * Lseq + tt) * (long)(2 * DIc) + d];
    }
    float s = 1.f / (1.f + __expf(-acc));
    acc *= s;
    xc[i] = acc;
    xcb[i] = (bf16_t)acc;
}

// selective scan: one block per batch, thread = channel d (DI=512).
// B_t/C_t (16+16 floats) broadcast through LDS each step.
__global__ __launch_bounds__(512)
void k_scan(const float* __restrict__ dt, const float* __restrict__ xz,
            const float* __restrict__ xc, const float* __restrict__ dbl,
            const float* __restrict__ A_log, const float* __restrict__ Dp,
            bf16_t* __restrict__ yact, int Lseq) {
    const int b = blockIdx.x, d = threadIdx.x;
    __shared__ float sB[16], sC[16];
    float a[16], h[16];
#pragma unroll
    for (int s = 0; s < 16; ++s) { a[s] = -__expf(A_log[d * 16 + s]); h[s] = 0.f; }
    const float Dd = Dp[d];
    for (int t = 0; t < Lseq; ++t) {
        long r = (long)b * Lseq + t;
        if (threadIdx.x < 32) {
            float v = dbl[r * 64 + 32 + threadIdx.x];
            if (threadIdx.x < 16) sB[threadIdx.x] = v; else sC[threadIdx.x - 16] = v;
        }
        __syncthreads();
        float dtv = dt[r * 512 + d];
        float xv  = xc[r * 512 + d];
        float zv  = xz[r * 1024 + 512 + d];
        float dtx = dtv * xv, y = 0.f;
#pragma unroll
        for (int s = 0; s < 16; ++s) {
            float dA = __expf(dtv * a[s]);
            h[s] = dA * h[s] + dtx * sB[s];
            y += h[s] * sC[s];
        }
        float sg = 1.f / (1.f + __expf(-zv));
        yact[r * 512 + d] = (bf16_t)((y + xv * Dd) * (zv * sg));
        __syncthreads();
    }
}

__global__ __launch_bounds__(256)
void k_layernorm(const float* __restrict__ in, const float* __restrict__ g,
                 const float* __restrict__ b,
                 float* __restrict__ outf, bf16_t* __restrict__ outb, int D) {
    __shared__ float red[256];
    long row = blockIdx.x;
    const float* x = in + row * (long)D;
    float s = 0.f;
    for (int i = threadIdx.x; i < D; i += 256) s += x[i];
    red[threadIdx.x] = s; __syncthreads();
    for (int off = 128; off > 0; off >>= 1) {
        if (threadIdx.x < off) red[threadIdx.x] += red[threadIdx.x + off];
        __syncthreads();
    }
    float mu = red[0] / D; __syncthreads();
    float v = 0.f;
    for (int i = threadIdx.x; i < D; i += 256) { float tt = x[i] - mu; v += tt * tt; }
    red[threadIdx.x] = v; __syncthreads();
    for (int off = 128; off > 0; off >>= 1) {
        if (threadIdx.x < off) red[threadIdx.x] += red[threadIdx.x + off];
        __syncthreads();
    }
    float inv = rsqrtf(red[0] / D + 1e-5f);
    for (int i = threadIdx.x; i < D; i += 256) {
        float o = (x[i] - mu) * inv * g[i] + b[i];
        if (outf) outf[row * (long)D + i] = o;
        if (outb) outb[row * (long)D + i] = (bf16_t)o;
    }
}

// ---------------------------------------------------------------------------
extern "C" void kernel_launch(void* const* d_in, const int* in_sizes, int n_in,
                              void* d_out, int out_size, void* d_ws, size_t ws_size,
                              hipStream_t stream) {
    (void)in_sizes; (void)n_in; (void)out_size; (void)ws_size;
    constexpr int B = 16, L = 1024, DM = 512, DSn = 16, KC = 4, E = 2, DFF = 2048, R = 32, DI = 512;
    const long M = (long)B * L;   // 16384 rows

    const float* x_in    = (const float*)d_in[0];
    const float* in_w    = (const float*)d_in[1];
    const float* conv_w  = (const float*)d_in[2];
    const float* conv_b  = (const float*)d_in[3];
    const float* xproj_w = (const float*)d_in[4];
    const float* dt_w    = (const float*)d_in[5];
    const float* dt_b    = (const float*)d_in[6];
    const float* A_log   = (const float*)d_in[7];
    const float* Dp      = (const float*)d_in[8];
    const float* out_w   = (const float*)d_in[9];
    const float* ff_w1   = (const float*)d_in[10];
    const float* ff_b1   = (const float*)d_in[11];
    const float* ff_w2   = (const float*)d_in[12];
    const float* ff_b2   = (const float*)d_in[13];
    const float* ln1_g   = (const float*)d_in[14];
    const float* ln1_b   = (const float*)d_in[15];
    const float* ln2_g   = (const float*)d_in[16];
    const float* ln2_b   = (const float*)d_in[17];
    const float* lnf_g   = (const float*)d_in[18];
    const float* lnf_b   = (const float*)d_in[19];

    char* wp = (char*)d_ws;
    auto alloc = [&](size_t bytes) -> void* {
        void* p = wp; wp += (bytes + 255) & ~((size_t)255); return p;
    };

    // bf16 weight copies (recomputed every launch: deterministic, no caching)
    bf16_t* inw_bf  = (bf16_t*)alloc(sizeof(bf16_t) * (size_t)E * 2 * (2 * DI) * DM);
    bf16_t* xpw_bf  = (bf16_t*)alloc(sizeof(bf16_t) * (size_t)E * 2 * (R + 2 * DSn) * DI);
    bf16_t* dtw_bf  = (bf16_t*)alloc(sizeof(bf16_t) * (size_t)E * 2 * DI * R);
    bf16_t* outw_bf = (bf16_t*)alloc(sizeof(bf16_t) * (size_t)E * 2 * DM * DI);
    bf16_t* fw1_bf  = (bf16_t*)alloc(sizeof(bf16_t) * (size_t)E * DFF * DM);
    bf16_t* fw2_bf  = (bf16_t*)alloc(sizeof(bf16_t) * (size_t)E * DM * DFF);

    float*  xa    = (float*)alloc(sizeof(float) * M * DM);          // running x
    float*  xb    = (float*)alloc(sizeof(float) * M * DM);          // residual accumulator
    bf16_t* xbf   = (bf16_t*)alloc(sizeof(bf16_t) * M * DM);
    float*  xz    = (float*)alloc(sizeof(float) * M * 2 * DI);
    float*  xc    = (float*)alloc(sizeof(float) * M * DI);
    bf16_t* xcbf  = (bf16_t*)alloc(sizeof(bf16_t) * M * DI);
    float*  dbl   = (float*)alloc(sizeof(float) * M * (R + 2 * DSn));
    bf16_t* dblbf = (bf16_t*)alloc(sizeof(bf16_t) * M * (R + 2 * DSn));
    float*  dtv   = (float*)alloc(sizeof(float) * M * DI);
    bf16_t* yact  = (bf16_t*)alloc(sizeof(bf16_t) * M * DI);
    float*  xn    = (float*)alloc(sizeof(float) * M * DM);
    bf16_t* xnbf  = (bf16_t*)alloc(sizeof(bf16_t) * M * DM);
    bf16_t* ffhbf = (bf16_t*)alloc(sizeof(bf16_t) * M * DFF);
    float*  tsum  = (float*)alloc(sizeof(float) * M * DM);

    auto cvt = [&](const float* src, bf16_t* dst, long n) {
        k_f32_to_bf16<<<(int)((n + 255) / 256), 256, 0, stream>>>(src, dst, n);
    };
    cvt(in_w,    inw_bf,  (long)E * 2 * (2 * DI) * DM);
    cvt(xproj_w, xpw_bf,  (long)E * 2 * (R + 2 * DSn) * DI);
    cvt(dt_w,    dtw_bf,  (long)E * 2 * DI * R);
    cvt(out_w,   outw_bf, (long)E * 2 * DM * DI);
    cvt(ff_w1,   fw1_bf,  (long)E * DFF * DM);
    cvt(ff_w2,   fw2_bf,  (long)E * DM * DFF);

    auto gemm = [&](const bf16_t* A, int lda, const bf16_t* W, int ldw,
                    const float* bias, float* outf, bf16_t* outb,
                    int N, int K, int act, int smode) {
        dim3 g(N / 64, (int)(M / 128)), blk(128);
        k_gemm_bf16<<<g, blk, 0, stream>>>(A, lda, W, ldw, bias, outf, outb,
                                           N, K, act, smode, L);
    };

    hipMemcpyAsync(xa, x_in, sizeof(float) * M * DM, hipMemcpyDeviceToDevice, stream);

    const long nMD  = M * (long)DM;
    const long nMDI = M * (long)DI;

    for (int e = 0; e < E; ++e) {
        // residual accumulator starts as x
        hipMemcpyAsync(xb, xa, sizeof(float) * M * DM, hipMemcpyDeviceToDevice, stream);

        for (int dir = 0; dir < 2; ++dir) {
            int ed = e * 2 + dir;
            if (dir == 0)
                k_f32_to_bf16<<<(int)((nMD + 255) / 256), 256, 0, stream>>>(xa, xbf, nMD);
            else
                k_rev_to_bf16<<<(int)((nMD + 255) / 256), 256, 0, stream>>>(xa, xbf, L, DM, nMD);

            // xz = x @ in_w^T            (N=1024, K=512)
            gemm(xbf, DM, inw_bf + (size_t)ed * 2 * DI * DM, DM,
                 nullptr, xz, nullptr, 2 * DI, DM, 0, 0);
            // depthwise conv + SiLU
            k_conv_silu<<<(int)((nMDI + 255) / 256), 256, 0, stream>>>(
                xz, conv_w + (size_t)ed * DI * KC, conv_b + (size_t)ed * DI,
                xc, xcbf, L, DI, KC, nMDI);
            // dbl = xc @ xproj^T         (N=64, K=512)
            gemm(xcbf, DI, xpw_bf + (size_t)ed * (R + 2 * DSn) * DI, DI,
                 nullptr, dbl, dblbf, R + 2 * DSn, DI, 0, 0);
            // dt = softplus(dbl[:, :R] @ dt_w^T + dt_b)   (N=512, K=32, lda=64)
            gemm(dblbf, R + 2 * DSn, dtw_bf + (size_t)ed * DI * R, R,
                 dt_b + (size_t)ed * DI, dtv, nullptr, DI, R, 2, 0);
            // selective scan -> gated activation (bf16)
            k_scan<<<B, DI, 0, stream>>>(dtv, xz, xc, dbl,
                                         A_log + (size_t)ed * DI * DSn,
                                         Dp + (size_t)ed * DI, yact, L);
            // x += y @ out_w^T  (backward branch: add with L reversed)
            gemm(yact, DI, outw_bf + (size_t)ed * DM * DI, DI,
                 nullptr, xb, nullptr, DM, DI, 0, dir ? 2 : 1);
        }

        // xn = LN1(x)
        k_layernorm<<<(int)M, 256, 0, stream>>>(xb, ln1_g + (size_t)e * DM,
                                                ln1_b + (size_t)e * DM, xn, xnbf, DM);
        // h = gelu(xn @ ff_w1^T + b1)    (N=2048, K=512)
        gemm(xnbf, DM, fw1_bf + (size_t)e * DFF * DM, DM,
             ff_b1 + (size_t)e * DFF, nullptr, ffhbf, DFF, DM, 1, 0);
        // tsum = xn + h @ ff_w2^T + b2   (N=512, K=2048)
        hipMemcpyAsync(tsum, xn, sizeof(float) * M * DM, hipMemcpyDeviceToDevice, stream);
        gemm(ffhbf, DFF, fw2_bf + (size_t)e * DM * DFF, DFF,
             ff_b2 + (size_t)e * DM, tsum, nullptr, DM, DFF, 0, 1);
        // x = LN2(tsum)
        k_layernorm<<<(int)M, 256, 0, stream>>>(tsum, ln2_g + (size_t)e * DM,
                                                ln2_b + (size_t)e * DM, xa, nullptr, DM);
    }

    // final LN -> d_out (f32)
    k_layernorm<<<(int)M, 256, 0, stream>>>(xa, lnf_g, lnf_b, (float*)d_out, nullptr, DM);
}
